// MultichannelMultiheadAttention_42537356100155
// MI455X (gfx1250) — compile-verified, matrix operand order checked by reading, not run on hardware
//
#include <hip/hip_runtime.h>

typedef __bf16 bf16_t;
typedef __attribute__((ext_vector_type(16))) __bf16 v16bf;
typedef __attribute__((ext_vector_type(8)))  float  v8f;

#define BDIM  2
#define CDIM  4
#define FDIM  512
#define WDIM  1024
#define NHEAD 8
#define DHEAD 64
#define NBC   (BDIM * CDIM)

// ---------- bf16 helpers (round-to-nearest-even) ----------
__device__ __forceinline__ bf16_t f2bf(float f) {
  union { float f; unsigned u; } a; a.f = f;
  unsigned r = a.u + 0x7FFFu + ((a.u >> 16) & 1u);
  unsigned short h = (unsigned short)(r >> 16);
  return __builtin_bit_cast(bf16_t, h);
}
__device__ __forceinline__ float bf2f(bf16_t b) {
  unsigned short h = __builtin_bit_cast(unsigned short, b);
  unsigned u = ((unsigned)h) << 16;
  return __builtin_bit_cast(float, u);
}

// ---------- CDNA5 async global->LDS copy (ASYNCcnt path), guarded ----------
#if defined(__gfx1250__) && __has_builtin(__builtin_amdgcn_global_load_async_to_lds_b128)
#define HAVE_ASYNC_LDS 1
#else
#define HAVE_ASYNC_LDS 0
#endif

typedef int v4i_b128 __attribute__((vector_size(16)));
typedef __attribute__((address_space(1))) v4i_b128* gas_b128_t;
typedef __attribute__((address_space(3))) v4i_b128* las_b128_t;

__device__ __forceinline__ void async_copy_b128(const bf16_t* gsrc, bf16_t* lds_dst) {
#if HAVE_ASYNC_LDS
  __builtin_amdgcn_global_load_async_to_lds_b128(
      (gas_b128_t)gsrc, (las_b128_t)lds_dst, 0, 0);
#else
  *(uint4*)lds_dst = *(const uint4*)gsrc;
#endif
}
__device__ __forceinline__ void async_drain() {
#if HAVE_ASYNC_LDS
  asm volatile("s_wait_asynccnt 0" ::: "memory");
#endif
}

// ---------- WMMA 16x16x32 bf16 fragment loader ----------
// Tile stored row-major [rows][ldk], K contiguous (ldk even).
// A-frag: row = M row.  B-frag: row = N column (B tile stored [N][K]).
// ISA 16-bit 16x32 layout: lanes 0-15 hold K {0..7,16..23}, lanes 16-31 hold
// K {8..15,24..31}; consecutive K pairs per dword -> aligned 32-bit LDS loads.
__device__ __forceinline__ v16bf load_frag16x32(const bf16_t* base, int row, int ldk, int lane) {
  const int kbase = (lane & 16) ? 8 : 0;
  union { v16bf v; unsigned u[8]; } r;
  const unsigned* p = (const unsigned*)(base + (size_t)row * ldk);
#pragma unroll
  for (int vi = 0; vi < 8; ++vi) {
    const int kk = (vi < 4) ? (kbase + 2 * vi) : (16 + kbase + 2 * (vi - 4));
    r.u[vi] = p[kk >> 1];
  }
  return r.v;
}

// ---------- f32 -> bf16 conversion ----------
__global__ void cvt_bf16_kernel(const float* __restrict__ in, bf16_t* __restrict__ out, int n) {
  int i = blockIdx.x * blockDim.x + threadIdx.x;
  if (i < n) out[i] = f2bf(in[i]);
}

// ---------- gate: g[b,c] = sum_h gA[c,h] * mean_w x[b,c,h,w] ----------
__global__ void gate_kernel(const float* __restrict__ x, const float* __restrict__ gA,
                            float* __restrict__ g) {
  __shared__ float red[256];
  const int bc = blockIdx.x, c = bc & (CDIM - 1);
  const float* xp = x + (size_t)bc * FDIM * WDIM;
  float acc = 0.f;
  for (int h = threadIdx.x; h < FDIM; h += blockDim.x) {
    const float* row = xp + (size_t)h * WDIM;
    float s = 0.f;
    for (int w = 0; w < WDIM; ++w) s += row[w];
    acc += (s * (1.f / (float)WDIM)) * gA[c * FDIM + h];
  }
  red[threadIdx.x] = acc;
  __syncthreads();
  for (int off = 128; off > 0; off >>= 1) {
    if ((int)threadIdx.x < off) red[threadIdx.x] += red[threadIdx.x + off];
    __syncthreads();
  }
  if (threadIdx.x == 0) g[bc] = red[0];
}

// ---------- per-channel GEMM: Y[bc] = W[c] (512x512) * X[bc] (512x1024) ----------
// Block tile 64(M) x 128(N): 8 waves in 2x4 grid, each wave a 2x2 set of 16x16
// WMMA tiles (4 wmma per K-step from 2A+2B fragments). A tile staged via
// async global->LDS b128; B tile transposed manually (global [K][N] -> LDS [N][K]).
template <int OUTF32>
__global__ void __launch_bounds__(256) mc_gemm_kernel(const bf16_t* __restrict__ Wt, // [C][F][F]
                                                      const bf16_t* __restrict__ X,  // [NBC][F][W]
                                                      bf16_t* __restrict__ Yb,
                                                      float* __restrict__ Yf) {
  __shared__ bf16_t As[64][32];  // [M][K]
  __shared__ bf16_t Bs[128][32]; // [N][K]
  const int tid = threadIdx.x, lane = tid & 31, wid = tid >> 5;
  const int wm = wid >> 2, wn = wid & 3;
  const int bc = blockIdx.z, c = bc & (CDIM - 1);
  const int mBase = blockIdx.y * 64, nBase = blockIdx.x * 128;
  const bf16_t* Wp = Wt + (size_t)c * FDIM * FDIM;
  const bf16_t* Xp = X + (size_t)bc * FDIM * WDIM;

  const int arow = tid >> 2, acol = (tid & 3) * 8;  // A stage: one b128 per thread
  const int kloc = tid >> 3, wg = (tid & 7) * 16;   // B stage: 16 bf16 per thread

  v8f acc00 = {}, acc01 = {}, acc10 = {}, acc11 = {};
  for (int k0 = 0; k0 < FDIM; k0 += 32) {
    if (k0) __syncthreads(); // previous compute done before LDS overwrite
    // A tile: straight copy -> async global->LDS
    async_copy_b128(Wp + (size_t)(mBase + arow) * FDIM + k0 + acol, &As[arow][acol]);
    // B tile: transpose stage (2-byte scatter, must go through VGPRs)
    {
      const bf16_t* src = Xp + (size_t)(k0 + kloc) * WDIM + nBase + wg;
      uint4 b0 = *(const uint4*)(src);
      uint4 b1 = *(const uint4*)(src + 8);
      const unsigned short* e0 = (const unsigned short*)&b0;
      const unsigned short* e1 = (const unsigned short*)&b1;
#pragma unroll
      for (int i = 0; i < 8; ++i) {
        *(unsigned short*)&Bs[wg + i][kloc]     = e0[i];
        *(unsigned short*)&Bs[wg + 8 + i][kloc] = e1[i];
      }
    }
    if (k0 + 32 < FDIM) { // emits global_prefetch_b8
      __builtin_prefetch(Wp + (size_t)(mBase + arow) * FDIM + (k0 + 32) + acol, 0, 3);
      __builtin_prefetch(Xp + (size_t)(k0 + 32 + kloc) * WDIM + nBase + wg, 0, 3);
    }
    async_drain();
    __syncthreads();
    v16bf a0 = load_frag16x32(&As[0][0], wm * 32 + 0  + (lane & 15), 32, lane);
    v16bf a1 = load_frag16x32(&As[0][0], wm * 32 + 16 + (lane & 15), 32, lane);
    v16bf b0 = load_frag16x32(&Bs[0][0], wn * 32 + 0  + (lane & 15), 32, lane);
    v16bf b1 = load_frag16x32(&Bs[0][0], wn * 32 + 16 + (lane & 15), 32, lane);
    acc00 = __builtin_amdgcn_wmma_f32_16x16x32_bf16(false, a0, false, b0, (short)0, acc00, false, false);
    acc01 = __builtin_amdgcn_wmma_f32_16x16x32_bf16(false, a0, false, b1, (short)0, acc01, false, false);
    acc10 = __builtin_amdgcn_wmma_f32_16x16x32_bf16(false, a1, false, b0, (short)0, acc10, false, false);
    acc11 = __builtin_amdgcn_wmma_f32_16x16x32_bf16(false, a1, false, b1, (short)0, acc11, false, false);
  }
  const int rbase = (lane & 16) ? 8 : 0;
  const int colA = nBase + wn * 32 + (lane & 15);
#pragma unroll
  for (int r = 0; r < 8; ++r) {
    const int row0 = mBase + wm * 32 + rbase + r;
    const int row1 = row0 + 16;
    const size_t base = (size_t)bc * FDIM * WDIM;
    if (OUTF32) {
      Yf[base + (size_t)row0 * WDIM + colA]      = acc00[r];
      Yf[base + (size_t)row0 * WDIM + colA + 16] = acc01[r];
      Yf[base + (size_t)row1 * WDIM + colA]      = acc10[r];
      Yf[base + (size_t)row1 * WDIM + colA + 16] = acc11[r];
    } else {
      Yb[base + (size_t)row0 * WDIM + colA]      = f2bf(acc00[r]);
      Yb[base + (size_t)row0 * WDIM + colA + 16] = f2bf(acc01[r]);
      Yb[base + (size_t)row1 * WDIM + colA]      = f2bf(acc10[r]);
      Yb[base + (size_t)row1 * WDIM + colA + 16] = f2bf(acc11[r]);
    }
  }
}

// ---------- depthwise conv (1,9) pad 4 + optional interleaved RoPE ----------
__global__ void convrope_kernel(const bf16_t* __restrict__ in, const float* __restrict__ dw,
                                bf16_t* __restrict__ out, int doRope) {
  const size_t gid = (size_t)blockIdx.x * blockDim.x + threadIdx.x;
  const size_t total = (size_t)NBC * FDIM * WDIM;
  if (gid >= total) return;
  const int w = (int)(gid % WDIM);
  const int h = (int)((gid / WDIM) % FDIM);
  const int bc = (int)(gid / ((size_t)WDIM * FDIM));
  const int c = bc & (CDIM - 1);
  const bf16_t* base = in + (size_t)bc * FDIM * WDIM;

  float self = 0.f;
#pragma unroll
  for (int j = 0; j < 9; ++j) {
    const int ww = w + j - 4;
    if (ww >= 0 && ww < WDIM) self += dw[c * 9 + j] * bf2f(base[(size_t)h * WDIM + ww]);
  }
  float val;
  if (doRope) {
    const int hp = h ^ 1; // rotation partner (same head, d index ^1)
    float pair = 0.f;
#pragma unroll
    for (int j = 0; j < 9; ++j) {
      const int ww = w + j - 4;
      if (ww >= 0 && ww < WDIM) pair += dw[c * 9 + j] * bf2f(base[(size_t)hp * WDIM + ww]);
    }
    const int dd = h & (DHEAD - 1);
    const float invf = __powf(10000.f, -(float)(dd & ~1) / (float)DHEAD);
    const float ang = (float)w * invf;
    float sn, cs;
    __sincosf(ang, &sn, &cs);
    val = (dd & 1) ? (self * cs + pair * sn) : (self * cs - pair * sn);
  } else {
    val = self;
  }
  out[gid] = f2bf(val);
}

// ---------- flash attention: one (b,c,head) x 64 queries per block ----------
__global__ void __launch_bounds__(256) attn_flash_kernel(const bf16_t* __restrict__ Q,
                                                         const bf16_t* __restrict__ K,
                                                         const bf16_t* __restrict__ V,
                                                         const float* __restrict__ G,
                                                         bf16_t* __restrict__ O) {
  __shared__ bf16_t Qs[64][64]; // [q][d]
  __shared__ bf16_t Ks[32][64]; // [key][d]   (B-frag layout for S)
  __shared__ bf16_t Vt[64][32]; // [d][key]   (B-frag layout for O; straight copy -> async)
  __shared__ bf16_t Ps[64][32]; // softmax probs, bf16
  __shared__ float  Ss[64][32]; // scaled scores
  __shared__ float  m_s[64], l_s[64], alpha_s[64], linv_s[64];

  const int tid = threadIdx.x, lane = tid & 31, wid = tid >> 5;
  const int swm = wid >> 1, swn = wid & 1; // 4x2 waves over 64q x 32k (and 64q x 2 d-halves)
  const int rbase = (lane & 16) ? 8 : 0;
  const int bc = blockIdx.z, head = blockIdx.y, qBase = blockIdx.x * 64;
  const size_t hoff = ((size_t)bc * FDIM + head * DHEAD) * WDIM;
  const bf16_t* Qp = Q + hoff;
  const bf16_t* Kp = K + hoff;
  const bf16_t* Vp = V + hoff;
  const float scale = G[bc] * 0.04419417382415922f; // 1/sqrt(512)

  // stage Q tile once (transpose global [d][w] -> LDS [q][d])
#pragma unroll
  for (int i = 0; i < 16; ++i) {
    const int idx = tid * 16 + i;
    const int dd = idx >> 6, q = idx & 63;
    Qs[q][dd] = Qp[(size_t)dd * WDIM + qBase + q];
  }
  if (tid < 64) { m_s[tid] = -1e30f; l_s[tid] = 0.f; }
  v8f oacc0 = {}, oacc1 = {};

  for (int kb = 0; kb < WDIM; kb += 32) {
    __syncthreads(); // Qs/init ready; previous iter's Vt/Ks reads done
    {
      const int dd = tid >> 2, w8 = (tid & 3) * 8;
      // V tile: straight copy -> async global->LDS
      async_copy_b128(Vp + (size_t)dd * WDIM + kb + w8, &Vt[dd][w8]);
      // K tile: transpose stage
      uint4 kv = *(const uint4*)(Kp + (size_t)dd * WDIM + kb + w8);
      const unsigned short* e = (const unsigned short*)&kv;
#pragma unroll
      for (int i = 0; i < 8; ++i) *(unsigned short*)&Ks[w8 + i][dd] = e[i];
    }
    async_drain();
    __syncthreads();
    // S = Q K^T over d=64 (two K=32 WMMA steps)
    v8f s = {};
    {
      v16bf a0 = load_frag16x32(&Qs[0][0],      swm * 16 + (lane & 15), 64, lane);
      v16bf b0 = load_frag16x32(&Ks[0][0],      swn * 16 + (lane & 15), 64, lane);
      s = __builtin_amdgcn_wmma_f32_16x16x32_bf16(false, a0, false, b0, (short)0, s, false, false);
      v16bf a1 = load_frag16x32(&Qs[0][0] + 32, swm * 16 + (lane & 15), 64, lane);
      v16bf b1 = load_frag16x32(&Ks[0][0] + 32, swn * 16 + (lane & 15), 64, lane);
      s = __builtin_amdgcn_wmma_f32_16x16x32_bf16(false, a1, false, b1, (short)0, s, false, false);
    }
#pragma unroll
    for (int r = 0; r < 8; ++r)
      Ss[swm * 16 + rbase + r][swn * 16 + (lane & 15)] = s[r] * scale;
    __syncthreads();
    // online softmax (one thread per query row)
    if (tid < 64) {
      const int row = tid;
      const float mold = m_s[row];
      float mx = mold;
#pragma unroll
      for (int j = 0; j < 32; ++j) mx = fmaxf(mx, Ss[row][j]);
      const float alpha = __expf(mold - mx);
      float sum = 0.f;
#pragma unroll
      for (int j = 0; j < 32; ++j) {
        const float p = __expf(Ss[row][j] - mx);
        Ps[row][j] = f2bf(p);
        sum += p;
      }
      m_s[row] = mx;
      l_s[row] = l_s[row] * alpha + sum;
      alpha_s[row] = alpha;
    }
    __syncthreads();
    // O = O*alpha + P V   (K=32 -> exactly one WMMA step per d-subtile)
    {
#pragma unroll
      for (int r = 0; r < 8; ++r) {
        const float a = alpha_s[swm * 16 + rbase + r];
        oacc0[r] *= a;
        oacc1[r] *= a;
      }
      v16bf p  = load_frag16x32(&Ps[0][0], swm * 16 + (lane & 15), 32, lane);
      v16bf v0 = load_frag16x32(&Vt[0][0], swn * 32 + 0  + (lane & 15), 32, lane);
      v16bf v1 = load_frag16x32(&Vt[0][0], swn * 32 + 16 + (lane & 15), 32, lane);
      oacc0 = __builtin_amdgcn_wmma_f32_16x16x32_bf16(false, p, false, v0, (short)0, oacc0, false, false);
      oacc1 = __builtin_amdgcn_wmma_f32_16x16x32_bf16(false, p, false, v1, (short)0, oacc1, false, false);
    }
  }
  __syncthreads();
  if (tid < 64) linv_s[tid] = 1.f / l_s[tid];
  __syncthreads();
#pragma unroll
  for (int r = 0; r < 8; ++r) {
    const int qrow = qBase + swm * 16 + rbase + r;
    const float inv = linv_s[swm * 16 + rbase + r];
    const int h0 = head * DHEAD + swn * 32 + (lane & 15);
    O[((size_t)bc * FDIM + h0)      * WDIM + qrow] = f2bf(oacc0[r] * inv);
    O[((size_t)bc * FDIM + h0 + 16) * WDIM + qrow] = f2bf(oacc1[r] * inv);
  }
}

// ---------- host side ----------
extern "C" void kernel_launch(void* const* d_in, const int* in_sizes, int n_in,
                              void* d_out, int out_size, void* d_ws, size_t ws_size,
                              hipStream_t stream) {
  (void)in_sizes; (void)n_in; (void)out_size; (void)ws_size;
  const float* x      = (const float*)d_in[0];
  const float* gA_pw  = (const float*)d_in[1];
  const float* q_pw   = (const float*)d_in[2];
  const float* q_dw   = (const float*)d_in[3];
  const float* k_pw   = (const float*)d_in[4];
  const float* k_dw   = (const float*)d_in[5];
  const float* v_pw   = (const float*)d_in[6];
  const float* v_dw   = (const float*)d_in[7];
  const float* out_pw = (const float*)d_in[8];
  float* out = (float*)d_out;

  const size_t NX = (size_t)NBC * FDIM * WDIM;  // 4,194,304
  const size_t NW = (size_t)CDIM * FDIM * FDIM; // 1,048,576

  char* ws = (char*)d_ws;
  size_t o = 0;
  bf16_t* Xb = (bf16_t*)(ws + o); o += NX * 2;
  bf16_t* Wq = (bf16_t*)(ws + o); o += NW * 2;
  bf16_t* Wk = (bf16_t*)(ws + o); o += NW * 2;
  bf16_t* Wv = (bf16_t*)(ws + o); o += NW * 2;
  bf16_t* Wo = (bf16_t*)(ws + o); o += NW * 2;
  bf16_t* Pq = (bf16_t*)(ws + o); o += NX * 2;
  bf16_t* Pk = (bf16_t*)(ws + o); o += NX * 2;
  bf16_t* Pv = (bf16_t*)(ws + o); o += NX * 2;
  bf16_t* Q2 = (bf16_t*)(ws + o); o += NX * 2;
  bf16_t* K2 = (bf16_t*)(ws + o); o += NX * 2;
  bf16_t* V2 = (bf16_t*)(ws + o); o += NX * 2;
  bf16_t* Oa = (bf16_t*)(ws + o); o += NX * 2;
  float*  Gv = (float*)(ws + o);  o += 256;

  const int CT = 256;
  cvt_bf16_kernel<<<(unsigned)((NX + CT - 1) / CT), CT, 0, stream>>>(x, Xb, (int)NX);
  cvt_bf16_kernel<<<(unsigned)((NW + CT - 1) / CT), CT, 0, stream>>>(q_pw, Wq, (int)NW);
  cvt_bf16_kernel<<<(unsigned)((NW + CT - 1) / CT), CT, 0, stream>>>(k_pw, Wk, (int)NW);
  cvt_bf16_kernel<<<(unsigned)((NW + CT - 1) / CT), CT, 0, stream>>>(v_pw, Wv, (int)NW);
  cvt_bf16_kernel<<<(unsigned)((NW + CT - 1) / CT), CT, 0, stream>>>(out_pw, Wo, (int)NW);
  gate_kernel<<<NBC, CT, 0, stream>>>(x, gA_pw, Gv);

  dim3 ggrid(WDIM / 128, FDIM / 64, NBC);
  mc_gemm_kernel<0><<<ggrid, CT, 0, stream>>>(Wq, Xb, Pq, nullptr);
  mc_gemm_kernel<0><<<ggrid, CT, 0, stream>>>(Wk, Xb, Pk, nullptr);
  mc_gemm_kernel<0><<<ggrid, CT, 0, stream>>>(Wv, Xb, Pv, nullptr);

  const unsigned cgrid = (unsigned)((NX + CT - 1) / CT);
  convrope_kernel<<<cgrid, CT, 0, stream>>>(Pq, q_dw, Q2, 1);
  convrope_kernel<<<cgrid, CT, 0, stream>>>(Pk, k_dw, K2, 1);
  convrope_kernel<<<cgrid, CT, 0, stream>>>(Pv, v_dw, V2, 0);

  dim3 agrid(WDIM / 64, NHEAD, NBC);
  attn_flash_kernel<<<agrid, CT, 0, stream>>>(Q2, K2, V2, Gv, Oa);

  mc_gemm_kernel<1><<<ggrid, CT, 0, stream>>>(Wo, Oa, nullptr, out);
}
